// SARNTN_61005715472760
// MI455X (gfx1250) — matrix operand order repaired and checked by reading, not compiled, not run on hardware
//
#include <hip/hip_runtime.h>
#include <hip/hip_bf16.h>

typedef __attribute__((ext_vector_type(16))) __bf16 v16bf;
typedef __attribute__((ext_vector_type(8)))  float  v8f;
typedef __attribute__((ext_vector_type(4)))  unsigned int u32x4;
typedef __attribute__((ext_vector_type(8)))  int i32x8;
typedef __attribute__((ext_vector_type(4)))  int i32x4;

#define BATCH 64
#define SEQL  512
#define HDIM  64
#define NCHUNK 516            /* 4 * 129 : 128 V-chunks + 1 W-chunk per q-group */
#define CHUNK_BF16 2048       /* 4 ntiles * 32 lanes * 16 bf16 per chunk (4KB) */
#define VPACK_ELEMS (NCHUNK * CHUNK_BF16)

// Tensor Data Mover availability (device pass only; host pass falls back).
#if defined(__has_builtin)
#  if __has_builtin(__builtin_amdgcn_tensor_load_to_lds) && \
      __has_builtin(__builtin_amdgcn_s_wait_tensorcnt)
#    define USE_TDM 1
#  endif
#endif
#ifndef USE_TDM
#  define USE_TDM 0
#endif

// ---------------------------------------------------------------------------
// Pack V (64x128x128) and W (128x64) into WMMA-B fragment order, bf16.
// Chunk pc = qi*129 + p; p==128 is the W-augmentation chunk for q-group qi.
// Within a chunk: [j(ntile 0..3)][lane 0..31][e 0..15],
// B-layout: column n = lane&15, K = (lane>>4)*16 + e.
// ---------------------------------------------------------------------------
__global__ void packV_kernel(const float* __restrict__ V,
                             const float* __restrict__ W,
                             __bf16* __restrict__ Vpack) {
  int idx = blockIdx.x * blockDim.x + threadIdx.x;
  if (idx >= VPACK_ELEMS) return;
  int e    = idx & 15;
  int lane = (idx >> 4) & 31;
  int j    = (idx >> 9) & 3;
  int pc   = idx >> 11;
  int p    = pc % 129;
  int qi   = pc / 129;
  int n  = lane & 15;
  int hi = lane >> 4;
  int kk = hi * 16 + e;
  int q  = qi * 32 + kk;       // 0..127
  int i  = j * 16 + n;         // output channel 0..63
  float v = (p < 128) ? V[(size_t)i * (128 * 128) + (size_t)p * 128 + q]
                      : W[(size_t)q * 64 + i];
  Vpack[idx] = (__bf16)v;
}

// ---------------------------------------------------------------------------
// Embedding gather -> bf16 node buffer (rows of 64).
// ---------------------------------------------------------------------------
__global__ void gather_kernel(const int* __restrict__ tokens,
                              const float* __restrict__ embed,
                              __bf16* __restrict__ out, int total) {
  int idx = blockIdx.x * blockDim.x + threadIdx.x;
  if (idx >= total) return;
  int r = idx >> 6;
  int h = idx & 63;
  int t = tokens[r];
  out[idx] = (__bf16)embed[(size_t)t * 64 + h];
}

#if USE_TDM
// Issue one TDM load: 4KB contiguous chunk of Vpack -> LDS (one op per wave).
// D# built per CDNA5 ISA ch.8: 1-D tile, data_size=8B, tile_dim0=512, type=2.
__device__ __forceinline__ void tdm_load_chunk(const __bf16* Vpack, int chunk,
                                               void* ldsDst) {
  unsigned lds = (unsigned)(size_t)ldsDst;  // generic ptr low 32 bits = LDS byte addr
  unsigned long long ga =
      (unsigned long long)(size_t)Vpack + (unsigned long long)chunk * 4096ull;
  const unsigned td0 = (unsigned)(NCHUNK * 512);  // tensor_dim0 in 8B units

  u32x4 g0;
  g0[0] = 1u;                                    // count=1, user descriptor
  g0[1] = lds;                                   // lds_addr
  g0[2] = (unsigned)(ga & 0xffffffffu);          // global_addr[31:0]
  g0[3] = (unsigned)(ga >> 32) | (2u << 30);     // global_addr[56:32] | type=2

  i32x8 g1;
  g1[0] = (int)(3u << 16);                       // data_size = 8B
  g1[1] = (int)((td0 & 0xffffu) << 16);          // tensor_dim0[15:0]
  g1[2] = (int)(td0 >> 16) | (1 << 16);          // tensor_dim0[31:16], tensor_dim1=1
  g1[3] = (int)(512u << 16);                     // tile_dim0 = 512
  g1[4] = 0;                                     // tile_dim1=0, tile_dim2=0
  g1[5] = (int)td0;                              // tensor_dim0_stride[31:0]
  g1[6] = 0;
  g1[7] = 0;

  i32x4 z4 = {0, 0, 0, 0};
#if __clang_major__ >= 23
  i32x8 z8 = {0, 0, 0, 0, 0, 0, 0, 0};
  __builtin_amdgcn_tensor_load_to_lds(g0, g1, z4, z4, z8, 0);
#else
  __builtin_amdgcn_tensor_load_to_lds(g0, g1, z4, z4, 0);
#endif
}
#endif

// ---------------------------------------------------------------------------
// One combine level:  out[r, 0:64] = tanh( X[r,:] @ Vaug + b )
// X[r,(p,q)] = c[r,p]*c[r,q] (plus unit-scaled c rows for the W part);
// c = input node buffer reinterpreted as (rows,128) bf16.
// Block = 256 thr = 8 waves; wave handles 16 rows x all 64 outputs.
// V chunks stream through LDS: TRIPLE buffered, TDM-fed, ONE barrier/chunk.
// ---------------------------------------------------------------------------
__global__ __launch_bounds__(256) void combine_kernel(
    const __bf16* __restrict__ cIn,    // rows x 128
    const __bf16* __restrict__ Vpack,  // NCHUNK * CHUNK_BF16
    const float*  __restrict__ bias,   // 64
    __bf16* __restrict__ outNodes,     // rows x 64
    int rows) {
  __shared__ __align__(16) __bf16 cTile[8][16][128];   // 32 KB
  __shared__ __align__(32) __bf16 Bst[3][CHUNK_BF16];  // 12 KB

  const int tid  = threadIdx.x;
  const int lane = tid & 31;
  const int m    = lane & 15;
  const int hi   = lane >> 4;
  // Scalarize wave-uniform values: guarantees s_cbranch (EXEC all-1s at WMMA).
  const int wave = __builtin_amdgcn_readfirstlane(tid >> 5);
  const int rowBase = blockIdx.x * 128 + wave * 16;
  const bool active = rowBase < rows;

  // Wave-private c tile: 16 rows x 128 bf16 = 4KB (contiguous in cIn).
  if (active) {
    const u32x4* src = (const u32x4*)(cIn + (size_t)rowBase * 128);
    u32x4* dst = (u32x4*)&cTile[wave][0][0];
#pragma unroll
    for (int k = 0; k < 8; ++k) dst[lane + 32 * k] = src[lane + 32 * k];
  }

  const u32x4* Vp4 = (const u32x4*)Vpack;      // 256 x 16B per chunk

  // Prologue: chunks 0 and 1 in flight.
#if USE_TDM
  if (wave == 0) {
    tdm_load_chunk(Vpack, 0, &Bst[0][0]);
    tdm_load_chunk(Vpack, 1, &Bst[1][0]);
  }
#else
  ((u32x4*)Bst[0])[tid] = Vp4[tid];
  ((u32x4*)Bst[1])[tid] = Vp4[256 + tid];
#endif

  v8f acc[4];
#pragma unroll
  for (int j = 0; j < 4; ++j) acc[j] = (v8f){};

  int chunk = 0;
  int bufc  = 0;                // buffer holding current chunk (rotates mod 3)
  for (int qi = 0; qi < 4; ++qi) {
    const int qb = qi * 32;
    // Loop-invariant A-side c slice for this q-group (fp32 for cheap scaling).
    float cvf[16];
    if (active) {
      union { u32x4 u[2]; v16bf v; } cvu;
      cvu.u[0] = *(const u32x4*)&cTile[wave][m][qb + hi * 16];
      cvu.u[1] = *(const u32x4*)&cTile[wave][m][qb + hi * 16 + 8];
#pragma unroll
      for (int e = 0; e < 16; ++e) cvf[e] = (float)cvu.v[e];
    }
    for (int p = 0; p <= 128; ++p, ++chunk) {
#if USE_TDM
      if (wave == 0) {
        // Outstanding TDM ops are a subset of {chunk, chunk+1}.
        if (chunk + 1 < NCHUNK) __builtin_amdgcn_s_wait_tensorcnt((short)1);
        else                    __builtin_amdgcn_s_wait_tensorcnt((short)0);
      }
#endif
      // Single barrier: publishes Bst[bufc] AND certifies the two-ahead
      // buffer (read in the previous iteration) is free for refill.
      __syncthreads();
      int buf2 = bufc + 2; if (buf2 >= 3) buf2 -= 3;
      if (chunk + 2 < NCHUNK) {
#if USE_TDM
        if (wave == 0) tdm_load_chunk(Vpack, chunk + 2, &Bst[buf2][0]);
#else
        ((u32x4*)Bst[buf2])[tid] = Vp4[(size_t)(chunk + 2) * 256 + tid];
        if (chunk + 3 < NCHUNK)
          __builtin_prefetch(&Vp4[(size_t)(chunk + 3) * 256 + tid], 0, 0);
#endif
      }
      if (active) {
        // Per-row scale: c[m,p] for V chunks, 1.0 for the W chunk (p==128).
        const float s = (p < 128) ? (float)cTile[wave][m][p] : 1.0f;
        v16bf a;
#pragma unroll
        for (int e = 0; e < 16; ++e) a[e] = (__bf16)(s * cvf[e]);
        // Preload all four B fragments, then issue WMMAs back-to-back.
        v16bf b0 = *(const v16bf*)&Bst[bufc][0 * 512 + lane * 16];
        v16bf b1 = *(const v16bf*)&Bst[bufc][1 * 512 + lane * 16];
        v16bf b2 = *(const v16bf*)&Bst[bufc][2 * 512 + lane * 16];
        v16bf b3 = *(const v16bf*)&Bst[bufc][3 * 512 + lane * 16];
        acc[0] = __builtin_amdgcn_wmma_f32_16x16x32_bf16(false, a, false, b0,
                                                         (short)0, acc[0], false, false);
        acc[1] = __builtin_amdgcn_wmma_f32_16x16x32_bf16(false, a, false, b1,
                                                         (short)0, acc[1], false, false);
        acc[2] = __builtin_amdgcn_wmma_f32_16x16x32_bf16(false, a, false, b2,
                                                         (short)0, acc[2], false, false);
        acc[3] = __builtin_amdgcn_wmma_f32_16x16x32_bf16(false, a, false, b3,
                                                         (short)0, acc[3], false, false);
      }
      bufc = (bufc + 1 == 3) ? 0 : bufc + 1;
    }
  }

  // Epilogue: D layout -> N = lane&15, M = hi*8 + vreg.
  if (active) {
#pragma unroll
    for (int j = 0; j < 4; ++j) {
      const int col = j * 16 + m;
      const float bb = bias[col];
#pragma unroll
      for (int v = 0; v < 8; ++v) {
        const int row = rowBase + hi * 8 + v;
        float val = tanhf(acc[j][v] + bb);
        outNodes[(size_t)row * 64 + col] = (__bf16)val;
      }
    }
  }
}

// ---------------------------------------------------------------------------
// Head: root(64x64 bf16) @ Wout(64x5) + bout, then log_softmax. fp32 out.
// ---------------------------------------------------------------------------
__global__ void head_kernel(const __bf16* __restrict__ root,
                            const float* __restrict__ Wout,
                            const float* __restrict__ bout,
                            float* __restrict__ out) {
  int b = blockIdx.x * blockDim.x + threadIdx.x;
  if (b >= BATCH) return;
  float lg[5];
#pragma unroll
  for (int o = 0; o < 5; ++o) {
    float s = bout[o];
    for (int h = 0; h < 64; ++h) s += (float)root[b * 64 + h] * Wout[h * 5 + o];
    lg[o] = s;
  }
  float mx = lg[0];
#pragma unroll
  for (int o = 1; o < 5; ++o) mx = fmaxf(mx, lg[o]);
  float se = 0.f;
#pragma unroll
  for (int o = 0; o < 5; ++o) se += expf(lg[o] - mx);
  float lse = mx + logf(se);
#pragma unroll
  for (int o = 0; o < 5; ++o) out[b * 5 + o] = lg[o] - lse;
}

// ---------------------------------------------------------------------------
extern "C" void kernel_launch(void* const* d_in, const int* in_sizes, int n_in,
                              void* d_out, int out_size, void* d_ws, size_t ws_size,
                              hipStream_t stream) {
  const int*   tokens = (const int*)d_in[0];
  const float* embed  = (const float*)d_in[1];
  const float* V      = (const float*)d_in[2];
  const float* W      = (const float*)d_in[3];
  const float* bvec   = (const float*)d_in[4];
  const float* Wout   = (const float*)d_in[5];
  const float* bout   = (const float*)d_in[6];

  char* ws = (char*)d_ws;
  const size_t vpackBytes = (size_t)VPACK_ELEMS * 2;          // ~2.02 MB
  const size_t nodeBytes  = (size_t)BATCH * SEQL * HDIM * 2;  // 4 MB
  __bf16* Vpack = (__bf16*)ws;
  __bf16* bufA  = (__bf16*)(ws + ((vpackBytes + 255) & ~(size_t)255));
  __bf16* bufB  = (__bf16*)((char*)bufA + nodeBytes);

  packV_kernel<<<(VPACK_ELEMS + 255) / 256, 256, 0, stream>>>(V, W, Vpack);

  const int gatherTotal = BATCH * SEQL * HDIM;
  gather_kernel<<<(gatherTotal + 255) / 256, 256, 0, stream>>>(tokens, embed,
                                                               bufA, gatherTotal);

  __bf16* cur = bufA;
  __bf16* nxt = bufB;
  for (int L = SEQL; L > 1; L >>= 1) {
    const int rows = BATCH * (L >> 1);       // c rows at this level
    const int blocks = (rows + 127) / 128;
    combine_kernel<<<blocks, 256, 0, stream>>>(cur, Vpack, bvec, nxt, rows);
    __bf16* t = cur; cur = nxt; nxt = t;
  }

  head_kernel<<<1, 64, 0, stream>>>(cur, Wout, bout, (float*)d_out);
}
